// MultiHeadAttention_13185549599123
// MI455X (gfx1250) — matrix operand order repaired and checked by reading, not compiled, not run on hardware
//
#include <hip/hip_runtime.h>
#include <hip/hip_bf16.h>

typedef __bf16 bf16;
typedef __attribute__((ext_vector_type(16))) __bf16 v16bf;
typedef __attribute__((ext_vector_type(8)))  float  v8f;

#define DEV __device__ __forceinline__

static constexpr int Bd  = 4;
static constexpr int Ld  = 2048;
static constexpr int Dd  = 1024;
static constexpr int Hh  = 16;
static constexpr int Hd  = 64;
static constexpr int BL  = Bd * Ld;        // 8192
static constexpr int D3  = 3 * Dd;         // 3072

DEV v8f wmma_bf16(v16bf a, v16bf b, v8f c) {
    // D = A(16x32 bf16) * B(32x16 bf16) + C(16x16 f32)
    return __builtin_amdgcn_wmma_f32_16x16x32_bf16(
        false, a, false, b, (short)0, c, false, false);
}

union Frag16 { v16bf v; uint4 q[2]; };

// A-matrix 16x32 bf16 fragment (ISA 7.12.2): lane m=lane&15 holds row m;
// element e -> K = (e&8)*2 + (lane>>4)*8 + (e&7)  => two contiguous 16B chunks.
DEV v16bf load_a_frag(const bf16* a, int ld, int row0, int k0, int lane) {
    int m = lane & 15, hi = lane >> 4;
    const bf16* p = a + (size_t)(row0 + m) * ld + k0;
    Frag16 u;
    u.q[0] = *reinterpret_cast<const uint4*>(p + hi * 8);
    u.q[1] = *reinterpret_cast<const uint4*>(p + 16 + hi * 8);
    return u.v;
}

// B-matrix 32x16 bf16 fragment: lane col n=lane&15; element e -> K = (e&8)*2 + (lane>>4)*8 + (e&7).
// bt is the K-major ("B-transposed") matrix: bt[n][k], ld = K stride.
DEV v16bf load_b_frag(const bf16* bt, int ld, int n0, int k0, int lane) {
    int n = lane & 15, hi = lane >> 4;
    const bf16* p = bt + (size_t)(n0 + n) * ld + k0;
    Frag16 u;
    u.q[0] = *reinterpret_cast<const uint4*>(p + hi * 8);
    u.q[1] = *reinterpret_cast<const uint4*>(p + 16 + hi * 8);
    return u.v;
}

// Async global -> LDS (CDNA5; ASYNCcnt-tracked, 16B per lane).
DEV void async_g2l_b128(unsigned lds_off, const bf16* gptr) {
    asm volatile("global_load_async_to_lds_b128 %0, %1, off"
                 :: "v"(lds_off), "v"(gptr) : "memory");
}

DEV float rowmax16(float x) {
    x = fmaxf(x, __shfl_xor(x, 1));
    x = fmaxf(x, __shfl_xor(x, 2));
    x = fmaxf(x, __shfl_xor(x, 4));
    x = fmaxf(x, __shfl_xor(x, 8));
    return x;
}
DEV float rowsum16(float x) {
    x += __shfl_xor(x, 1);
    x += __shfl_xor(x, 2);
    x += __shfl_xor(x, 4);
    x += __shfl_xor(x, 8);
    return x;
}

// ---------------------------------------------------------------- converts
__global__ void cvt_f32_bf16(const float* __restrict__ src, bf16* __restrict__ dst, int n) {
    int i = blockIdx.x * blockDim.x + threadIdx.x;
    if (i < n) dst[i] = (bf16)src[i];
}

// src [R][C] f32 -> dst [C][R] bf16
__global__ void cvt_transpose(const float* __restrict__ src, bf16* __restrict__ dst,
                              int R, int C) {
    int i = blockIdx.x * blockDim.x + threadIdx.x;
    if (i < R * C) {
        int r = i / C, c = i - r * C;
        dst[(size_t)c * R + r] = (bf16)src[i];
    }
}

// ---------------------------------------------------------------- GEMM core
// 128 threads (4 waves). Block tile 128(M) x 64(N), K stepped by 64 per
// staged buffer (two 32-K WMMA substeps). B tile (64 x 64, 8KB) is
// double-buffered in LDS via async global->LDS copies shared by all 4 waves;
// A fragments load direct (per-wave-exclusive rows).
DEV void gemm_core(const bf16* __restrict__ A, const bf16* __restrict__ Bt,
                   int m0 /*this wave's 32-row base*/, int n0, v8f acc[2][4]) {
    __shared__ __align__(32) bf16 ldsB[2 * 64 * 64];   // 16 KiB, two 8 KiB buffers
    const int KK = Dd;
    const int nk = KK / 64;                            // 16 staged batches
    int tid  = threadIdx.x;
    int lane = tid & 31;

    // Each of the 128 threads copies 64 bytes of the 8 KiB B tile:
    // row = tid>>1 (0..63); half = tid&1 selects a 32-element (64B) chunk.
    const bf16* gB = Bt + (size_t)(n0 + (tid >> 1)) * KK + (tid & 1) * 32;
    const unsigned lbase = (unsigned)(uintptr_t)&ldsB[(tid >> 1) * 64 + (tid & 1) * 32];

    auto issue = [&](int kOfs, unsigned lOfs) {
        async_g2l_b128(lbase + lOfs,       gB + kOfs);
        async_g2l_b128(lbase + lOfs + 16u, gB + kOfs + 8);
        async_g2l_b128(lbase + lOfs + 32u, gB + kOfs + 16);
        async_g2l_b128(lbase + lOfs + 48u, gB + kOfs + 24);
    };

    issue(0, 0);                                       // prologue: batch 0
    for (int kb = 0; kb < nk; ++kb) {
        // Always prefetch the next batch (clamped on the final iteration; it
        // rewrites the buffer that is not being read and drains at s_endpgm).
        int kIss = (kb + 1 < nk) ? (kb + 1) * 64 : kb * 64;
        issue(kIss, (unsigned)(((kb + 1) & 1) << 13));
        // 4 instrs of the new batch outstanding; async loads complete in
        // order, so <=4 outstanding means batch kb has landed in LDS.
        asm volatile("s_wait_asynccnt 4" ::: "memory");
        __syncthreads();                               // publish LDS to all waves

        const bf16* Bl = &ldsB[(kb & 1) * 4096];
#pragma unroll
        for (int ks = 0; ks < 2; ++ks) {
            int k0 = kb * 64 + ks * 32;
            v16bf a0 = load_a_frag(A, KK, m0,      k0, lane);
            v16bf a1 = load_a_frag(A, KK, m0 + 16, k0, lane);
#pragma unroll
            for (int t = 0; t < 4; ++t) {
                v16bf bf = load_b_frag(Bl, 64, t * 16, ks * 32, lane);
                acc[0][t] = wmma_bf16(a0, bf, acc[0][t]);
                acc[1][t] = wmma_bf16(a1, bf, acc[1][t]);
            }
        }
        __syncthreads();                               // reads done before overwrite
    }
}

// ---------------------------------------------------------------- QKV GEMM
// C[BL, 3072] = A[BL,1024](bf16) x Bt[3072,1024]^T + b_in, scattered to
// Q[B,H,L,64], K[B,H,L,64], Vt[B,H,64,L]  (all bf16).
__global__ void __launch_bounds__(128, 1)
gemm_qkv(const bf16* __restrict__ A, const bf16* __restrict__ Bt,
         const float* __restrict__ bias,
         bf16* __restrict__ Q, bf16* __restrict__ Kc, bf16* __restrict__ Vt) {
    int lane = threadIdx.x & 31, wave = threadIdx.x >> 5;
    int m0 = blockIdx.x * 128 + wave * 32;
    int n0 = blockIdx.y * 64;
    int hi = lane >> 4, col = lane & 15;

    v8f zero = {};
    v8f acc[2][4];
#pragma unroll
    for (int i = 0; i < 2; ++i)
#pragma unroll
        for (int t = 0; t < 4; ++t) acc[i][t] = zero;

    gemm_core(A, Bt, m0, n0, acc);

#pragma unroll
    for (int mt = 0; mt < 2; ++mt) {
#pragma unroll
        for (int t = 0; t < 4; ++t) {
            int n    = n0 + t * 16 + col;
            float bv = bias[n];
            int part = n >> 10;           // 0=Q 1=K 2=V
            int rem  = n & 1023;
            int h    = rem >> 6;
            int hd   = rem & 63;
#pragma unroll
            for (int r = 0; r < 8; ++r) {
                int tok = m0 + mt * 16 + hi * 8 + r;
                int b   = tok >> 11;          // /2048
                int l   = tok & 2047;
                bf16 v  = (bf16)(acc[mt][t][r] + bv);
                size_t bh = (size_t)(b * Hh + h);
                if      (part == 0) Q [(bh * Ld + l) * Hd + hd] = v;
                else if (part == 1) Kc[(bh * Ld + l) * Hd + hd] = v;
                else                Vt[(bh * Hd + hd) * Ld + l] = v;
            }
        }
    }
}

// ---------------------------------------------------------------- attention
// One wave (32 threads) per (b,h, 16-query tile). Online softmax over 64-wide
// KV blocks (rescale work amortized over 2x the KV per update).
// Q,K: [bh][L][64] bf16; Vt: [bh][64][L] bf16.
__global__ void __launch_bounds__(32, 1)
attn_fwd(const bf16* __restrict__ Q, const bf16* __restrict__ Kc,
         const bf16* __restrict__ Vt, bf16* __restrict__ att_out) {
    __shared__ __align__(32) bf16 p_lds[16 * 64];      // P tile 16x64, 2 KiB

    int lane  = threadIdx.x;
    int bh    = blockIdx.x >> 7;      // L/16 = 128 query tiles
    int qt    = blockIdx.x & 127;
    int qbase = qt * 16;
    int hi    = lane >> 4, col = lane & 15;

    const bf16* Qb = Q  + (size_t)bh * Ld * Hd;
    const bf16* Kb = Kc + (size_t)bh * Ld * Hd;
    const bf16* Vb = Vt + (size_t)bh * Hd * Ld;

    // Q fragments for K-steps hd=[0,32) and [32,64), reused every KV block.
    v16bf qf0 = load_a_frag(Qb, Hd, qbase, 0,  lane);
    v16bf qf1 = load_a_frag(Qb, Hd, qbase, 32, lane);

    v8f zero = {};
    v8f O[4];
#pragma unroll
    for (int t = 0; t < 4; ++t) O[t] = zero;
    float m[8], lsum[8];
#pragma unroll
    for (int r = 0; r < 8; ++r) { m[r] = -3.0e38f; lsum[r] = 0.0f; }

    const float scale = 0.125f;       // 1/sqrt(64)
    int nblocks = (qbase + 16 + 63) >> 6;

    for (int kb = 0; kb < nblocks; ++kb) {
        int cbase = kb * 64;
        // S = Q K^T : four 16x16 N-tiles, K=64 via two k-steps each.
        v8f s[4];
#pragma unroll
        for (int j = 0; j < 4; ++j) {
            s[j] = zero;
            s[j] = wmma_bf16(qf0, load_b_frag(Kb, Hd, cbase + j * 16, 0,  lane), s[j]);
            s[j] = wmma_bf16(qf1, load_b_frag(Kb, Hd, cbase + j * 16, 32, lane), s[j]);
        }

#pragma unroll
        for (int r = 0; r < 8; ++r) {
            int qi = qbase + hi * 8 + r;
            float v[4];
#pragma unroll
            for (int j = 0; j < 4; ++j)
                v[j] = (cbase + j * 16 + col <= qi) ? s[j][r] * scale : -3.0e38f;
            float mn = rowmax16(fmaxf(fmaxf(v[0], v[1]), fmaxf(v[2], v[3])));
            mn = fmaxf(mn, m[r]);
            float alpha = __expf(m[r] - mn);
            m[r] = mn;
            lsum[r] *= alpha;
#pragma unroll
            for (int t = 0; t < 4; ++t) O[t][r] *= alpha;
            float p[4], psum = 0.0f;
#pragma unroll
            for (int j = 0; j < 4; ++j) { p[j] = __expf(v[j] - mn); psum += p[j]; }
            lsum[r] += rowsum16(psum);
#pragma unroll
            for (int j = 0; j < 4; ++j)
                p_lds[(hi * 8 + r) * 64 + j * 16 + col] = (bf16)p[j];
        }
        __syncthreads();
        // P as two A-fragments (16x32 each), accumulate O += P * V.
        v16bf pf0 = load_a_frag(p_lds, 64, 0, 0,  lane);
        v16bf pf1 = load_a_frag(p_lds, 64, 0, 32, lane);
        __syncthreads();
#pragma unroll
        for (int t = 0; t < 4; ++t) {
            O[t] = wmma_bf16(pf0, load_b_frag(Vb, Ld, t * 16, cbase,      lane), O[t]);
            O[t] = wmma_bf16(pf1, load_b_frag(Vb, Ld, t * 16, cbase + 32, lane), O[t]);
        }
    }

    int b = bh >> 4, h = bh & 15;
#pragma unroll
    for (int t = 0; t < 4; ++t)
#pragma unroll
        for (int r = 0; r < 8; ++r) {
            float o = O[t][r] / lsum[r];
            size_t tok = (size_t)b * Ld + qbase + hi * 8 + r;
            att_out[tok * Dd + h * 64 + t * 16 + col] = (bf16)o;
        }
}

// ---------------------------------------------------------------- out proj
// out[BL,1024] f32 = att_out[BL,1024](bf16) x w_out_t[1024,1024]^T + b_out
__global__ void __launch_bounds__(128, 1)
gemm_out(const bf16* __restrict__ A, const bf16* __restrict__ Bt,
         const float* __restrict__ bias, float* __restrict__ out) {
    int lane = threadIdx.x & 31, wave = threadIdx.x >> 5;
    int m0 = blockIdx.x * 128 + wave * 32;
    int n0 = blockIdx.y * 64;
    int hi = lane >> 4, col = lane & 15;

    v8f zero = {};
    v8f acc[2][4];
#pragma unroll
    for (int i = 0; i < 2; ++i)
#pragma unroll
        for (int t = 0; t < 4; ++t) acc[i][t] = zero;

    gemm_core(A, Bt, m0, n0, acc);

#pragma unroll
    for (int mt = 0; mt < 2; ++mt)
#pragma unroll
        for (int t = 0; t < 4; ++t) {
            int n    = n0 + t * 16 + col;
            float bv = bias[n];
#pragma unroll
            for (int r = 0; r < 8; ++r) {
                int row = m0 + mt * 16 + hi * 8 + r;
                out[(size_t)row * Dd + n] = acc[mt][t][r] + bv;
            }
        }
}

// ---------------------------------------------------------------- launch
extern "C" void kernel_launch(void* const* d_in, const int* in_sizes, int n_in,
                              void* d_out, int out_size, void* d_ws, size_t ws_size,
                              hipStream_t stream) {
    const float* x     = (const float*)d_in[0];   // [4,2048,1024]
    const float* w_in  = (const float*)d_in[1];   // [1024,3072]
    const float* b_in  = (const float*)d_in[2];   // [3072]
    const float* w_out = (const float*)d_in[3];   // [1024,1024]
    const float* b_out = (const float*)d_in[4];   // [1024]
    float* out = (float*)d_out;                   // [4,2048,1024]

    // workspace layout (bf16 buffers), ~88 MiB total
    const size_t MB = 1024 * 1024;
    char* ws = (char*)d_ws;
    bf16* xb    = (bf16*)(ws);              // 16 MiB  [8192,1024]
    bf16* wint  = (bf16*)(ws + 16 * MB);    //  6 MiB  [3072,1024]  (w_in^T)
    bf16* woutt = (bf16*)(ws + 22 * MB);    //  2 MiB  [1024,1024]  (w_out^T)
    bf16* Qb    = (bf16*)(ws + 24 * MB);    // 16 MiB  [64,2048,64]
    bf16* Kb    = (bf16*)(ws + 40 * MB);    // 16 MiB  [64,2048,64]
    bf16* Vt    = (bf16*)(ws + 56 * MB);    // 16 MiB  [64,64,2048]
    bf16* attb  = (bf16*)(ws + 72 * MB);    // 16 MiB  [8192,1024]

    // 1) conversions
    {
        int n = BL * Dd;
        cvt_f32_bf16<<<(n + 255) / 256, 256, 0, stream>>>(x, xb, n);
        int n1 = Dd * D3;
        cvt_transpose<<<(n1 + 255) / 256, 256, 0, stream>>>(w_in, wint, Dd, D3);
        int n2 = Dd * Dd;
        cvt_transpose<<<(n2 + 255) / 256, 256, 0, stream>>>(w_out, woutt, Dd, Dd);
    }
    // 2) QKV projection + scatter
    gemm_qkv<<<dim3(BL / 128, D3 / 64), 128, 0, stream>>>(xb, wint, b_in, Qb, Kb, Vt);
    // 3) causal flash attention, one wave per (b,h,q-tile)
    attn_fwd<<<Bd * Hh * (Ld / 16), 32, 0, stream>>>(Qb, Kb, Vt, attb);
    // 4) output projection
    gemm_out<<<dim3(BL / 128, Dd / 64), 128, 0, stream>>>(attb, woutt, b_out, out);
}
